// Attention_6640019440503
// MI455X (gfx1250) — matrix operand order repaired and checked by reading, not compiled, not run on hardware
//
#include <hip/hip_runtime.h>
#include <hip/hip_bf16.h>

// ---------------------------------------------------------------------------
// MI455X (gfx1250) multi-head attention: B=2, N=4096, C=512, H=8, HD=64.
// All matmuls run on v_wmma_f32_16x16x32_f16 (f16 in, f32 accumulate).
// Kernel 3 stages its A tile with global_load_async_to_lds_b128 (ASYNCcnt).
// ---------------------------------------------------------------------------

typedef _Float16 v16h __attribute__((ext_vector_type(16)));
typedef _Float16 v8h  __attribute__((ext_vector_type(8)));
typedef float    v8f  __attribute__((ext_vector_type(8)));

union FragH { v16h v; v8h h[2]; };

__device__ __forceinline__ v8f wmma_f16(v16h a, v16h b, v8f c) {
  // D = A(16x32 f16) * B(32x16 f16) + C(16x16 f32)
  return __builtin_amdgcn_wmma_f32_16x16x32_f16(
      /*neg_a=*/false, a, /*neg_b=*/false, b,
      /*c_mod=*/(short)0, c, /*reuse_a=*/false, /*reuse_b=*/false);
}

constexpr int kB    = 2;
constexpr int kN    = 4096;
constexpr int kC    = 512;
constexpr int kH    = 8;
constexpr int kHD   = 64;
constexpr int kM    = kB * kN;        // 8192 rows of x
constexpr int kN3   = 3 * kC;         // 1536 qkv cols
constexpr float kScale = 0.125f;      // 64^-0.5
constexpr int kLS = 40;               // LDS row stride: 32 f16 + 8 pad (16B align)

// ---------------------------------------------------------------------------
// Kernel 1: QKV = x @ w_qkv -> Q (scaled, [b,h,n,64] f16), K ([b,h,n,64] f16),
// V stored TRANSPOSED ([b,h,64,n] f16) so attention never transposes.
// Block tile 256x64 (8 waves), wave tile 32x64, K-step 32: 8 WMMA / wave / step
// reusing 4 B-fragments. Q/K/V select is uniform per block (n0 multiple of 64).
// ---------------------------------------------------------------------------
__global__ __launch_bounds__(256) void qkv_gemm(
    const float* __restrict__ x, const float* __restrict__ wqkv,
    _Float16* __restrict__ Qh, _Float16* __restrict__ Kh,
    _Float16* __restrict__ Vt) {
  __shared__ _Float16 lA[256 * kLS];   // A tile [256][32] f16
  __shared__ _Float16 lB[64 * kLS];    // B tile TRANSPOSED [n=64][k=32] f16

  const int tid  = threadIdx.x;
  const int lane = tid & 31;
  const int wave = tid >> 5;
  const int lr   = lane & 15;
  const int hf   = lane >> 4;
  const int m0   = blockIdx.x * 256;
  const int n0   = blockIdx.y * 64;

  const int brow = tid >> 3, bc0 = (tid & 7) * 8;

  v8f acc[2][4] = {};

  for (int kk = 0; kk < kC; kk += 32) {
    // Stage A: one 512-col row per thread, 32 f32 -> f16, packed b128 stores
    {
      const float* ga = x + (size_t)(m0 + tid) * kC + kk;
      _Float16* la = lA + tid * kLS;
      #pragma unroll
      for (int c = 0; c < 4; ++c) {
        v8h t;
        #pragma unroll
        for (int i = 0; i < 8; ++i) t[i] = (_Float16)ga[c * 8 + i];
        *(v8h*)(la + c * 8) = t;
      }
    }
    // Stage B transposed: w_qkv[kk+k][n0+n] -> lB[n][k]
    {
      const float* gb = wqkv + (size_t)(kk + brow) * kN3 + n0 + bc0;
      #pragma unroll
      for (int j = 0; j < 8; ++j)
        lB[(bc0 + j) * kLS + brow] = (_Float16)gb[j];
    }
    __syncthreads();

    // Batch all fragment loads, then burst 8 WMMAs
    FragH a[2];
    #pragma unroll
    for (int mt = 0; mt < 2; ++mt) {
      const _Float16* ap = lA + (wave * 32 + mt * 16 + lr) * kLS + hf * 8;
      a[mt].h[0] = *(const v8h*)(ap);
      a[mt].h[1] = *(const v8h*)(ap + 16);
    }
    FragH bf[4];
    #pragma unroll
    for (int t = 0; t < 4; ++t) {
      const _Float16* bp = lB + (t * 16 + lr) * kLS + hf * 8;
      bf[t].h[0] = *(const v8h*)(bp);
      bf[t].h[1] = *(const v8h*)(bp + 16);
    }
    #pragma unroll
    for (int mt = 0; mt < 2; ++mt)
      #pragma unroll
      for (int t = 0; t < 4; ++t)
        acc[mt][t] = wmma_f16(a[mt].v, bf[t].v, acc[mt][t]);
    __syncthreads();
  }

  // Epilogue: Q/K/V select is UNIFORM per block -> single scalar branch
  const int three = n0 >> 9;          // 0=Q 1=K 2=V
  const int nc0   = n0 & 511;
  const int mbase = m0 + wave * 32 + hf * 8;
  if (three == 2) {
    #pragma unroll
    for (int mt = 0; mt < 2; ++mt) {
      #pragma unroll
      for (int t = 0; t < 4; ++t) {
        const int col  = nc0 + t * 16 + lr;
        const int head = col >> 6, hd = col & 63;
        #pragma unroll
        for (int r = 0; r < 8; ++r) {
          const int gm = mbase + mt * 16 + r;
          const int b = gm >> 12, nq = gm & 4095;
          Vt[((size_t)(b * kH + head) * kHD + hd) * kN + nq] =
              (_Float16)acc[mt][t][r];
        }
      }
    }
  } else {
    _Float16* __restrict__ dst = three ? Kh : Qh;
    const float sc = three ? 1.f : kScale;
    #pragma unroll
    for (int mt = 0; mt < 2; ++mt) {
      #pragma unroll
      for (int t = 0; t < 4; ++t) {
        const int col  = nc0 + t * 16 + lr;
        const int head = col >> 6, hd = col & 63;
        #pragma unroll
        for (int r = 0; r < 8; ++r) {
          const int gm = mbase + mt * 16 + r;
          const int b = gm >> 12, nq = gm & 4095;
          dst[((size_t)((b * kH + head) << 12) + nq) * kHD + hd] =
              (_Float16)(acc[mt][t][r] * sc);
        }
      }
    }
  }
}

// ---------------------------------------------------------------------------
// Kernel 2: flash attention per (b,h). Each wave owns 16 q-rows.
// S^T = K_j(16x64) @ Q^T(64x16)  -> acc lanes hold one q column each, so
// softmax max/sum = 8 in-lane ops + one shfl_xor(16). P^T accumulators map
// (one cross-half shuffle) directly onto the B-fragment of O^T += V^T @ P^T.
// ---------------------------------------------------------------------------
__global__ __launch_bounds__(256) void flash_attn(
    const _Float16* __restrict__ Qh, const _Float16* __restrict__ Kh,
    const _Float16* __restrict__ Vt, _Float16* __restrict__ AO) {
  const int tid  = threadIdx.x;
  const int lane = tid & 31;
  const int wave = tid >> 5;
  const int lr   = lane & 15;
  const int hf   = lane >> 4;
  const int bh   = blockIdx.y;
  const int q0   = blockIdx.x * 128 + wave * 16;

  const _Float16* Qb = Qh + (size_t)bh * kN * kHD;
  const _Float16* Kb = Kh + (size_t)bh * kN * kHD;
  const _Float16* Vb = Vt + (size_t)bh * kHD * kN;

  // Q^T B-fragments (persistent across all kv steps)
  FragH qb[2];
  #pragma unroll
  for (int c = 0; c < 2; ++c) {
    const _Float16* qp = Qb + (size_t)(q0 + lr) * kHD + c * 32 + hf * 8;
    qb[c].h[0] = *(const v8h*)(qp);
    qb[c].h[1] = *(const v8h*)(qp + 16);
  }

  v8f o[4] = {};                 // O^T: 4 tiles of (16 hd x 16 q)
  float mrun = -1e30f, lrun = 0.f;

  for (int j = 0; j < kN; j += 32) {
    __builtin_prefetch(Kb + (size_t)(j + 32 + lr) * kHD, 0, 1);

    // S^T tiles: kv rows [j..j+15] and [j+16..j+31]
    v8f s0 = {}, s1 = {};
    #pragma unroll
    for (int c = 0; c < 2; ++c) {
      FragH ka;
      const _Float16* kp0 = Kb + (size_t)(j + lr) * kHD + c * 32 + hf * 8;
      ka.h[0] = *(const v8h*)(kp0);
      ka.h[1] = *(const v8h*)(kp0 + 16);
      s0 = wmma_f16(ka.v, qb[c].v, s0);
      const _Float16* kp1 = Kb + (size_t)(j + 16 + lr) * kHD + c * 32 + hf * 8;
      ka.h[0] = *(const v8h*)(kp1);
      ka.h[1] = *(const v8h*)(kp1 + 16);
      s1 = wmma_f16(ka.v, qb[c].v, s1);
    }

    // Online softmax; each lane owns one q column (replicated across halves)
    float mx = mrun;
    #pragma unroll
    for (int r = 0; r < 8; ++r) mx = fmaxf(mx, fmaxf(s0[r], s1[r]));
    mx = fmaxf(mx, __shfl_xor(mx, 16, 32));
    const float corr = __expf(mrun - mx);
    mrun = mx;

    float p0[8], p1[8], psum = 0.f;
    #pragma unroll
    for (int r = 0; r < 8; ++r) {
      p0[r] = __expf(s0[r] - mrun);
      p1[r] = __expf(s1[r] - mrun);
      psum += p0[r] + p1[r];
    }
    psum += __shfl_xor(psum, 16, 32);
    lrun = lrun * corr + psum;

    #pragma unroll
    for (int a = 0; a < 4; ++a) {
      #pragma unroll
      for (int r = 0; r < 8; ++r) o[a][r] *= corr;
    }

    // P^T B-fragment: lane<16 needs kv {0..15}, lane>=16 needs kv {16..31};
    // the other half of each 16-row tile lives in the partner lane.
    FragH pb;
    #pragma unroll
    for (int r = 0; r < 8; ++r) {
      const float x0 = __shfl_xor(p0[r], 16, 32);
      const float x1 = __shfl_xor(p1[r], 16, 32);
      pb.v[r]     = (_Float16)(hf ? x1 : p0[r]);
      pb.v[8 + r] = (_Float16)(hf ? p1[r] : x0);
    }

    // O^T += V^T @ P^T   (V pre-transposed: contiguous kv per hd row)
    #pragma unroll
    for (int a = 0; a < 4; ++a) {
      FragH va;
      const _Float16* vp = Vb + (size_t)(a * 16 + lr) * kN + j + hf * 16;
      va.h[0] = *(const v8h*)(vp);
      va.h[1] = *(const v8h*)(vp + 8);
      o[a] = wmma_f16(va.v, pb.v, o[a]);
    }
  }

  const float inv = 1.f / lrun;
  const int b = bh >> 3, head = bh & 7;
  #pragma unroll
  for (int a = 0; a < 4; ++a) {
    #pragma unroll
    for (int r = 0; r < 8; ++r) {
      const int hd = a * 16 + hf * 8 + r;
      const int q  = q0 + lr;
      AO[((size_t)(b * kN + q)) * kC + head * kHD + hd] =
          (_Float16)(o[a][r] * inv);
    }
  }
}

// ---------------------------------------------------------------------------
// Kernel 3: out = AttnO @ w_proj + b_proj (f16 A, f32 out). Tiling as kernel 1.
// A tile is f16->f16 (no conversion), so it is staged with the CDNA5 async
// global->LDS DMA path (global_load_async_to_lds_b128, ASYNCcnt).
// ---------------------------------------------------------------------------
__global__ __launch_bounds__(256) void proj_gemm(
    const _Float16* __restrict__ A, const float* __restrict__ W,
    const float* __restrict__ bias, float* __restrict__ out) {
  __shared__ _Float16 lA[256 * kLS];
  __shared__ _Float16 lB[64 * kLS];

  const int tid  = threadIdx.x;
  const int lane = tid & 31;
  const int wave = tid >> 5;
  const int lr   = lane & 15;
  const int hf   = lane >> 4;
  const int m0   = blockIdx.x * 256;
  const int n0   = blockIdx.y * 64;

  const int brow = tid >> 3, bc0 = (tid & 7) * 8;

  // LDS byte address of this thread's A-tile row (ISA 10.2: low 32 bits of
  // the generic pointer are the LDS address).
  const unsigned ldsRow = (unsigned)(uintptr_t)(lA + tid * kLS);

  v8f acc[2][4] = {};

  for (int kk = 0; kk < kC; kk += 32) {
    {
      // Async DMA: 4 x 16B per lane, global row -> LDS row. The signed
      // IOFFSET is added to BOTH the global and the LDS address, so the
      // within-row offsets line up on both sides.
      const unsigned long long gsrc =
          (unsigned long long)(uintptr_t)(A + (size_t)(m0 + tid) * kC + kk);
      asm volatile(
          "global_load_async_to_lds_b128 %0, %1, off\n\t"
          "global_load_async_to_lds_b128 %0, %1, off offset:16\n\t"
          "global_load_async_to_lds_b128 %0, %1, off offset:32\n\t"
          "global_load_async_to_lds_b128 %0, %1, off offset:48"
          :: "v"(ldsRow), "v"(gsrc) : "memory");
    }
    {
      const float* gb = W + (size_t)(kk + brow) * kC + n0 + bc0;
      #pragma unroll
      for (int j = 0; j < 8; ++j)
        lB[(bc0 + j) * kLS + brow] = (_Float16)gb[j];
    }
    asm volatile("s_wait_asynccnt 0" ::: "memory");
    __syncthreads();

    FragH a[2];
    #pragma unroll
    for (int mt = 0; mt < 2; ++mt) {
      const _Float16* ap = lA + (wave * 32 + mt * 16 + lr) * kLS + hf * 8;
      a[mt].h[0] = *(const v8h*)(ap);
      a[mt].h[1] = *(const v8h*)(ap + 16);
    }
    FragH bf[4];
    #pragma unroll
    for (int t = 0; t < 4; ++t) {
      const _Float16* bp = lB + (t * 16 + lr) * kLS + hf * 8;
      bf[t].h[0] = *(const v8h*)(bp);
      bf[t].h[1] = *(const v8h*)(bp + 16);
    }
    #pragma unroll
    for (int mt = 0; mt < 2; ++mt)
      #pragma unroll
      for (int t = 0; t < 4; ++t)
        acc[mt][t] = wmma_f16(a[mt].v, bf[t].v, acc[mt][t]);
    __syncthreads();
  }

  const int mbase = m0 + wave * 32 + hf * 8;
  #pragma unroll
  for (int mt = 0; mt < 2; ++mt) {
    #pragma unroll
    for (int t = 0; t < 4; ++t) {
      const int gn = n0 + t * 16 + lr;
      const float bb = bias[gn];
      #pragma unroll
      for (int r = 0; r < 8; ++r)
        out[(size_t)(mbase + mt * 16 + r) * kC + gn] = acc[mt][t][r] + bb;
    }
  }
}

// ---------------------------------------------------------------------------
extern "C" void kernel_launch(void* const* d_in, const int* in_sizes, int n_in,
                              void* d_out, int out_size, void* d_ws,
                              size_t ws_size, hipStream_t stream) {
  const float* x     = (const float*)d_in[0];   // [2,4096,512]
  const float* wqkv  = (const float*)d_in[1];   // [512,1536]
  const float* wproj = (const float*)d_in[2];   // [512,512]
  const float* bproj = (const float*)d_in[3];   // [512]
  float* out = (float*)d_out;

  char* ws = (char*)d_ws;
  _Float16* Qh = (_Float16*)(ws);                         // 8 MiB
  _Float16* Kh = (_Float16*)(ws + (size_t)8  * 1048576);  // 8 MiB
  _Float16* Vt = (_Float16*)(ws + (size_t)16 * 1048576);  // 8 MiB
  _Float16* AO = (_Float16*)(ws + (size_t)24 * 1048576);  // 8 MiB

  qkv_gemm  <<<dim3(kM / 256, kN3 / 64), 256, 0, stream>>>(x, wqkv, Qh, Kh, Vt);
  flash_attn<<<dim3(kN / 128, kB * kH),  256, 0, stream>>>(Qh, Kh, Vt, AO);
  proj_gemm <<<dim3(kM / 256, kC / 64),  256, 0, stream>>>(AO, wproj, bproj, out);
}
